// GATV2_64141041599030
// MI455X (gfx1250) — compile-verified, matrix operand order checked by reading, not run on hardware
//
#include <hip/hip_runtime.h>
#include <hip/hip_bf16.h>

typedef __attribute__((ext_vector_type(16))) _Float16 v16h;
typedef __attribute__((ext_vector_type(8)))  float    v8f;
typedef __attribute__((ext_vector_type(2)))  _Float16 v2h;

#define N_NODES 100000
#define N_EDGES 800000
#define N_ETOT  900000           // edges + self loops
#define NTILES  (N_NODES / 16)   // 6250 row tiles of 16

// ---- monotonic float<->uint mapping for atomic max on f32 ----
__device__ __forceinline__ unsigned fenc(float f) {
  unsigned b = __float_as_uint(f);
  return (b & 0x80000000u) ? ~b : (b | 0x80000000u);
}
__device__ __forceinline__ float fdec(unsigned u) {
  return (u & 0x80000000u) ? __uint_as_float(u & 0x7fffffffu)
                           : __uint_as_float(~u);
}

// =====================================================================
// Generic GEMM: C[N,64] = A[N,kdim] @ W[kdim,64] (+bias), f16 WMMA,
// f32 accumulate. 8 waves/block, each wave computes a 16x64 tile.
// =====================================================================
__global__ void __launch_bounds__(256)
gemm_wmma(const float* __restrict__ A, const float* __restrict__ W,
          const float* __restrict__ bias, float* __restrict__ C, int kdim)
{
  __shared__ _Float16 wlds[64 * 264];          // W transposed [n][k], padded
  const int tid    = threadIdx.x;
  const int lane   = tid & 31;
  const int wave   = tid >> 5;
  const int stride = kdim + 8;                 // pad to dodge bank conflicts

  // stage W -> LDS as f16, transposed to make (k,k+1) pairs contiguous
  for (int idx = tid; idx < kdim * 64; idx += 256) {
    int k = idx >> 6, n = idx & 63;
    wlds[n * stride + k] = (_Float16)W[idx];
  }
  __syncthreads();

  const int tile = blockIdx.x * 8 + wave;
  if (tile >= NTILES) return;
  const int row0 = tile * 16;
  const int m    = lane & 15;
  const int kb   = (lane & 16) ? 8 : 0;        // ISA 16-bit A-frag K base

  v8f acc[4] = {};
  const int ktiles = kdim >> 5;
  for (int kt = 0; kt < ktiles; ++kt) {
    // ---- A fragment: 16x32 f16, per ISA layout ----
    v16h a;
    const float* arow = A + (long long)(row0 + m) * kdim + kt * 32 + kb;
#pragma unroll
    for (int j = 0; j < 8; ++j) {
      int jk = 2 * j + ((j >= 4) ? 8 : 0);     // K 0..7 then 16..23 (+kb)
      float2 f = *reinterpret_cast<const float2*>(arow + jk);
      a[2 * j]     = (_Float16)f.x;
      a[2 * j + 1] = (_Float16)f.y;
    }
    // ---- 4 B fragments (32x16 each) from LDS ----
#pragma unroll
    for (int nt = 0; nt < 4; ++nt) {
      v16h b;
      const _Float16* brow = &wlds[(nt * 16 + m) * stride + kt * 32 + kb];
#pragma unroll
      for (int j = 0; j < 8; ++j) {
        int jk = 2 * j + ((j >= 4) ? 8 : 0);
        v2h h2 = *reinterpret_cast<const v2h*>(brow + jk);
        b[2 * j]     = h2[0];
        b[2 * j + 1] = h2[1];
      }
      acc[nt] = __builtin_amdgcn_wmma_f32_16x16x32_f16(
          false, a, false, b, (short)0, acc[nt], false, false);
    }
  }

  // ---- epilogue: D layout -> rows (lane<16: M=r, lane>=16: M=r+8) ----
  const int mrow = row0 + ((lane & 16) ? 8 : 0);
#pragma unroll
  for (int nt = 0; nt < 4; ++nt) {
    int   ncol = nt * 16 + m;
    float bv   = bias ? bias[ncol] : 0.0f;
#pragma unroll
    for (int r = 0; r < 8; ++r)
      C[(long long)(mrow + r) * 64 + ncol] = acc[nt][r] + bv;
  }
}

// =====================================================================
// Per-layer init: zero accumulators / softmax stats
// =====================================================================
__global__ void layer_init(float* accum, float* denom, unsigned* mmax)
{
  int t = blockIdx.x * blockDim.x + threadIdx.x;
  if (t < N_NODES * 64) accum[t] = 0.0f;
  if (t < N_NODES * 4) { denom[t] = 0.0f; mmax[t] = 0u; }
}

// =====================================================================
// Pass 1: attention scores + segment max (atomic max via uint encoding)
// one thread per (edge, head)
// =====================================================================
__global__ void edge_score(const float* __restrict__ xl, const float* __restrict__ xr,
                           const float* __restrict__ att, const int* __restrict__ ei,
                           float* __restrict__ score, unsigned* __restrict__ mmax)
{
  int t = blockIdx.x * blockDim.x + threadIdx.x;
  if (t >= N_ETOT * 4) return;
  int e = t >> 2, h = t & 3;
  int s = (e < N_EDGES) ? ei[e] : (e - N_EDGES);
  int d = (e < N_EDGES) ? ei[N_EDGES + e] : (e - N_EDGES);
  const float4* ps = reinterpret_cast<const float4*>(xl + (long long)s * 64 + h * 16);
  const float4* pd = reinterpret_cast<const float4*>(xr + (long long)d * 64 + h * 16);
  const float4* pa = reinterpret_cast<const float4*>(att + h * 16);
  float sc = 0.0f;
#pragma unroll
  for (int i = 0; i < 4; ++i) {
    float4 a = ps[i], b = pd[i], w = pa[i];
    float v;
    v = a.x + b.x; v = v > 0.0f ? v : 0.2f * v; sc += w.x * v;
    v = a.y + b.y; v = v > 0.0f ? v : 0.2f * v; sc += w.y * v;
    v = a.z + b.z; v = v > 0.0f ? v : 0.2f * v; sc += w.z * v;
    v = a.w + b.w; v = v > 0.0f ? v : 0.2f * v; sc += w.w * v;
  }
  score[t] = sc;
  atomicMax(&mmax[d * 4 + h], fenc(sc));
}

// =====================================================================
// Pass 2: ex = exp(score - max[dst]); denom[dst] += ex
// =====================================================================
__global__ void edge_exp(float* __restrict__ score, const unsigned* __restrict__ mmax,
                         float* __restrict__ denom, const int* __restrict__ ei)
{
  int t = blockIdx.x * blockDim.x + threadIdx.x;
  if (t >= N_ETOT * 4) return;
  int e = t >> 2, h = t & 3;
  int d = (e < N_EDGES) ? ei[N_EDGES + e] : (e - N_EDGES);
  float ex = __expf(score[t] - fdec(mmax[d * 4 + h]));
  score[t] = ex;                               // reuse buffer for ex
  atomicAdd(&denom[d * 4 + h], ex);
}

// =====================================================================
// Pass 3: accum[dst] += alpha * xl[src]; one lane per (edge, channel)
// so the 64 atomics of an edge are lane-contiguous.
// =====================================================================
__global__ void edge_scatter(const float* __restrict__ ex, const float* __restrict__ denom,
                             const float* __restrict__ xl, const int* __restrict__ ei,
                             float* __restrict__ accum)
{
  long long t = (long long)blockIdx.x * blockDim.x + threadIdx.x;
  if (t >= (long long)N_ETOT * 64) return;
  int e = (int)(t >> 6), c = (int)(t & 63), h = c >> 4;
  int s = (e < N_EDGES) ? ei[e] : (e - N_EDGES);
  int d = (e < N_EDGES) ? ei[N_EDGES + e] : (e - N_EDGES);
  float alpha = ex[e * 4 + h] / denom[d * 4 + h];
  atomicAdd(&accum[(long long)d * 64 + c], alpha * xl[(long long)s * 64 + c]);
}

// =====================================================================
// Node epilogue: out = elu(accum + conv_b)
// =====================================================================
__global__ void node_finish(const float* __restrict__ accum, const float* __restrict__ cb,
                            float* __restrict__ out)
{
  int t = blockIdx.x * blockDim.x + threadIdx.x;
  if (t >= N_NODES * 64) return;
  float v = accum[t] + cb[t & 63];
  out[t] = v > 0.0f ? v : (__expf(v) - 1.0f);
}

// =====================================================================
// log_softmax over 64 columns, one wave32 per node (2 cols/lane)
// =====================================================================
__global__ void __launch_bounds__(256)
log_softmax_rows(const float* __restrict__ logits, float* __restrict__ out)
{
  int wid = threadIdx.x >> 5, lane = threadIdx.x & 31;
  int node = blockIdx.x * 8 + wid;
  if (node >= N_NODES) return;
  const float* r = logits + (long long)node * 64;
  float v0 = r[lane], v1 = r[lane + 32];
  float mx = fmaxf(v0, v1);
#pragma unroll
  for (int off = 16; off >= 1; off >>= 1) mx = fmaxf(mx, __shfl_xor(mx, off, 32));
  float s = __expf(v0 - mx) + __expf(v1 - mx);
#pragma unroll
  for (int off = 16; off >= 1; off >>= 1) s += __shfl_xor(s, off, 32);
  float lse = mx + __logf(s);
  float* o = out + (long long)node * 64;
  o[lane] = v0 - lse;
  o[lane + 32] = v1 - lse;
}

// =====================================================================
extern "C" void kernel_launch(void* const* d_in, const int* in_sizes, int n_in,
                              void* d_out, int out_size, void* d_ws, size_t ws_size,
                              hipStream_t stream)
{
  (void)in_sizes; (void)n_in; (void)out_size; (void)ws_size;
  const float* x    = (const float*)d_in[0];   // [100000,256]
  const float* fc0w = (const float*)d_in[1];   // [256,64]
  const float* fc0b = (const float*)d_in[2];   // [64]
  const float* Wl   = (const float*)d_in[3];   // [2,64,64]
  const float* Wr   = (const float*)d_in[4];   // [2,64,64]
  const float* att  = (const float*)d_in[5];   // [2,64]
  const float* cb   = (const float*)d_in[6];   // [2,64]
  const float* fc1w = (const float*)d_in[7];   // [64,64]
  const float* fc1b = (const float*)d_in[8];   // [64]
  const int*   ei   = (const int*)d_in[9];     // [2,800000]

  float* ws    = (float*)d_ws;
  float* featA = ws;
  float* featB = featA + (size_t)N_NODES * 64;
  float* xl    = featB + (size_t)N_NODES * 64;
  float* xr    = xl    + (size_t)N_NODES * 64;
  float* accum = xr    + (size_t)N_NODES * 64;
  float* score = accum + (size_t)N_NODES * 64;   // [Etot,4], reused as ex
  float* denom = score + (size_t)N_ETOT  * 4;    // [N,4]
  unsigned* mmax = (unsigned*)(denom + (size_t)N_NODES * 4); // [N,4]

  const int GB  = (NTILES + 7) / 8;                       // gemm blocks
  const int NB  = (N_NODES * 64 + 255) / 256;             // node-channel blocks
  const int EB  = (N_ETOT * 4 + 255) / 256;               // edge-head blocks
  const int SB  = (int)(((long long)N_ETOT * 64 + 255) / 256); // scatter blocks
  const int LB  = (N_NODES + 7) / 8;                      // log-softmax blocks

  // h = x @ fc0_w + fc0_b
  gemm_wmma<<<GB, 256, 0, stream>>>(x, fc0w, fc0b, featA, 256);

  float* fin = featA; float* fout = featB;
  for (int k = 0; k < 2; ++k) {
    gemm_wmma<<<GB, 256, 0, stream>>>(fin, Wl + k * 64 * 64, nullptr, xl, 64);
    gemm_wmma<<<GB, 256, 0, stream>>>(fin, Wr + k * 64 * 64, nullptr, xr, 64);
    layer_init  <<<NB, 256, 0, stream>>>(accum, denom, mmax);
    edge_score  <<<EB, 256, 0, stream>>>(xl, xr, att + k * 64, ei, score, mmax);
    edge_exp    <<<EB, 256, 0, stream>>>(score, mmax, denom, ei);
    edge_scatter<<<SB, 256, 0, stream>>>(score, denom, xl, ei, accum);
    node_finish <<<NB, 256, 0, stream>>>(accum, cb + k * 64, fout);
    float* tmp = fin; fin = fout; fout = tmp;
  }

  // logits = last @ fc1_w + fc1_b  (xl is free again, use as logits buffer)
  gemm_wmma<<<GB, 256, 0, stream>>>(fin, fc1w, fc1b, xl, 64);
  log_softmax_rows<<<LB, 256, 0, stream>>>(xl, (float*)d_out);
}
